// EncoderLayer_50216757625212
// MI455X (gfx1250) — matrix operand order repaired and checked by reading, not compiled, not run on hardware
//
#include <hip/hip_runtime.h>
#include <hip/hip_bf16.h>

// ---------------------------------------------------------------------------
// Shapes (fixed by the reference)
// ---------------------------------------------------------------------------
#define BB   4
#define SS_  1024
#define DD   1024
#define HH   16
#define HD_  64
#define DFF_ 4096
#define MTOT (BB * SS_)                    // 4096 token rows
#define OUT2_ELEMS ((long)BB * SS_ * DD)   // 4,194,304

typedef __attribute__((ext_vector_type(16))) __bf16 v16bf;
typedef __attribute__((ext_vector_type(8)))  float  v8f;

union Frag32B {
    uint4          u[2];
    unsigned short s[16];
    __bf16         b[16];
    v16bf          v;
};

// Native bf16 convert (RNE) -> lets clang use the gfx1250 cvt ops.
__device__ __forceinline__ unsigned short f2bf(float f) {
    union { __bf16 h; unsigned short s; } u;
    u.h = (__bf16)f;
    return u.s;
}

// A fragment, 16x32 bf16 (ISA 7.12.2): per lane, elems 0..7 -> K = k0+8*half+i,
// elems 8..15 -> K = k0+16+8*half+(i-8).  Ap points at A[row] + k0.
__device__ __forceinline__ v16bf load_a_bf16(const unsigned short* __restrict__ Ap, int half) {
    Frag32B f;
    f.u[0] = *(const uint4*)(Ap + half * 8);
    f.u[1] = *(const uint4*)(Ap + 16 + half * 8);
    return f.v;
}

// Same fragment but source is f32 (converted on the fly with native cvt).
__device__ __forceinline__ v16bf load_a_f32(const float* __restrict__ Ap, int half) {
    Frag32B f;
    const float4 x0 = *(const float4*)(Ap + half * 8);
    const float4 x1 = *(const float4*)(Ap + half * 8 + 4);
    const float4 x2 = *(const float4*)(Ap + 16 + half * 8);
    const float4 x3 = *(const float4*)(Ap + 16 + half * 8 + 4);
    f.b[0]  = (__bf16)x0.x; f.b[1]  = (__bf16)x0.y; f.b[2]  = (__bf16)x0.z; f.b[3]  = (__bf16)x0.w;
    f.b[4]  = (__bf16)x1.x; f.b[5]  = (__bf16)x1.y; f.b[6]  = (__bf16)x1.z; f.b[7]  = (__bf16)x1.w;
    f.b[8]  = (__bf16)x2.x; f.b[9]  = (__bf16)x2.y; f.b[10] = (__bf16)x2.z; f.b[11] = (__bf16)x2.w;
    f.b[12] = (__bf16)x3.x; f.b[13] = (__bf16)x3.y; f.b[14] = (__bf16)x3.z; f.b[15] = (__bf16)x3.w;
    return f.v;
}

// B fragment, 32x16 bf16: per lane (col n = lane%16), elems 0..7 -> K = k0+8*half+i,
// elems 8..15 -> K = k0+16+8*half+(i-8).  Btp points at Bt[col] + k0 (Bt stored [N][K]).
__device__ __forceinline__ v16bf load_bt(const unsigned short* __restrict__ Btp, int half) {
    Frag32B f;
    f.u[0] = *(const uint4*)(Btp + half * 8);
    f.u[1] = *(const uint4*)(Btp + 16 + half * 8);
    return f.v;
}

// ---------------------------------------------------------------------------
// Generic WMMA GEMM:  C[M,N] = A[M,K] * Bt[N,K]^T  (+ epilogue)
// 256 threads = 8 waves; block tile 256x64; wave tile 32x64
// (2 A-frags x 4 B-frags -> 8 WMMAs per 32-wide K step), K double-buffered.
// M%256==0, N%64==0, K%32==0 (true for every GEMM here).
// ---------------------------------------------------------------------------
template <bool A_F32, bool HAS_BIAS, bool HAS_RES, bool RELU, bool MASKSCALE,
          bool OUT_F32, bool OUT_BF16, bool HEADA, bool HEADB, bool HEADC>
__global__ __launch_bounds__(256) void gemm_wmma(
    const void* __restrict__ Ap, const unsigned short* __restrict__ Bt,
    float* __restrict__ outF, unsigned short* __restrict__ outB,
    const float* __restrict__ bias, const float* __restrict__ res,
    const int* __restrict__ mask, float scale,
    int M, int N, int K, int lda, int ldbt, int ldc,
    long strideA, long strideB, long strideC, int headDiv)
{
    const int z    = blockIdx.z;
    const int lane = threadIdx.x & 31;
    const int wave = threadIdx.x >> 5;
    const int half = lane >> 4;
    const int r    = lane & 15;
    const int m0   = blockIdx.y * 256 + wave * 32;
    const int n0   = blockIdx.x * 64;

    long aoff, boff, coff;
    if (HEADA) aoff = (long)(z / headDiv) * M * lda + (long)(z % headDiv) * K;
    else       aoff = (long)z * strideA;
    if (HEADB) boff = (long)(z / headDiv) * N * ldbt + (long)(z % headDiv) * K;
    else       boff = (long)z * strideB;
    if (HEADC) coff = (long)(z / headDiv) * M * ldc + (long)(z % headDiv) * N;
    else       coff = (long)z * strideC;

    const unsigned short* Abf = (const unsigned short*)Ap;
    const float*          Af  = (const float*)Ap;

    const long arow0 = aoff + (long)(m0 + r) * lda;
    const long arow1 = aoff + (long)(m0 + 16 + r) * lda;
    const long brow  = boff + (long)(n0 + r) * ldbt;

    v8f acc[2][4] = {};
    v16bf a0, a1, b[4];

    // prologue: fragments for k = 0
    if (A_F32) { a0 = load_a_f32(Af + arow0, half);  a1 = load_a_f32(Af + arow1, half); }
    else       { a0 = load_a_bf16(Abf + arow0, half); a1 = load_a_bf16(Abf + arow1, half); }
#pragma unroll
    for (int nt = 0; nt < 4; ++nt) b[nt] = load_bt(Bt + brow + (long)nt * 16 * ldbt, half);

    // pipelined main loop: issue k+32 loads before the 8 WMMAs of k
    for (int k0 = 0; k0 + 32 < K; k0 += 32) {
        const int k1 = k0 + 32;
        v16bf na0, na1, nb[4];
        if (A_F32) { na0 = load_a_f32(Af + arow0 + k1, half);  na1 = load_a_f32(Af + arow1 + k1, half); }
        else       { na0 = load_a_bf16(Abf + arow0 + k1, half); na1 = load_a_bf16(Abf + arow1 + k1, half); }
#pragma unroll
        for (int nt = 0; nt < 4; ++nt) nb[nt] = load_bt(Bt + brow + (long)nt * 16 * ldbt + k1, half);

#pragma unroll
        for (int nt = 0; nt < 4; ++nt) {
            acc[0][nt] = __builtin_amdgcn_wmma_f32_16x16x32_bf16(
                false, a0, false, b[nt], (short)0, acc[0][nt], false, false);
            acc[1][nt] = __builtin_amdgcn_wmma_f32_16x16x32_bf16(
                false, a1, false, b[nt], (short)0, acc[1][nt], false, false);
        }
        a0 = na0; a1 = na1;
#pragma unroll
        for (int nt = 0; nt < 4; ++nt) b[nt] = nb[nt];
    }
    // epilogue K step
#pragma unroll
    for (int nt = 0; nt < 4; ++nt) {
        acc[0][nt] = __builtin_amdgcn_wmma_f32_16x16x32_bf16(
            false, a0, false, b[nt], (short)0, acc[0][nt], false, false);
        acc[1][nt] = __builtin_amdgcn_wmma_f32_16x16x32_bf16(
            false, a1, false, b[nt], (short)0, acc[1][nt], false, false);
    }

    const int* mrow = nullptr;
    if (MASKSCALE) mrow = mask + (long)(z / headDiv) * M * N;

#pragma unroll
    for (int mi = 0; mi < 2; ++mi) {
#pragma unroll
        for (int nt = 0; nt < 4; ++nt) {
#pragma unroll
            for (int j = 0; j < 8; ++j) {
                const int m = m0 + mi * 16 + half * 8 + j;   // ISA 7.12.2 C layout
                const int n = n0 + nt * 16 + r;
                float v = acc[mi][nt][j];
                if (HAS_BIAS)  v += bias[n];
                if (HAS_RES)   v += res[coff + (long)m * ldc + n];
                if (RELU)      v = fmaxf(v, 0.0f);
                if (MASKSCALE) { v *= scale; if (mrow[(long)m * N + n] == 0) v = -1e9f; }
                const long ci = coff + (long)m * ldc + n;
                if (OUT_F32)  outF[ci] = v;
                if (OUT_BF16) outB[ci] = f2bf(v);
            }
        }
    }
}

// ---------------------------------------------------------------------------
// Convert + transpose weight:  Wt[n][k] = bf16(W[k][n])
// ---------------------------------------------------------------------------
__global__ __launch_bounds__(256) void k_conv_transpose(
    const float* __restrict__ in, unsigned short* __restrict__ out, int K, int N)
{
    long idx = (long)blockIdx.x * 256 + threadIdx.x;
    if (idx >= (long)K * N) return;
    int  k = (int)(idx % K);
    long n = idx / K;
    out[idx] = f2bf(in[(long)k * N + n]);
}

// Build V^T (per head):  Vt[(b*H+h)*HD*S + d*S + s] = Pbf[b*S*D + s*D + h*HD + d]
__global__ __launch_bounds__(256) void k_build_vt(
    const unsigned short* __restrict__ Pbf, unsigned short* __restrict__ Vt)
{
    long idx = (long)blockIdx.x * 256 + threadIdx.x;      // total = B*H*HD*S
    if (idx >= (long)BB * HH * HD_ * SS_) return;
    int s  = (int)(idx & (SS_ - 1));
    int d  = (int)((idx >> 10) & (HD_ - 1));
    int bh = (int)(idx >> 16);
    int b  = bh >> 4, h = bh & 15;
    Vt[idx] = Pbf[(((long)b * SS_ + s) << 10) + h * HD_ + d];
}

// Row softmax over 1024 columns, in place (one block per row).
__global__ __launch_bounds__(256) void k_softmax(float* __restrict__ p)
{
    float4* pr = (float4*)(p + ((long)blockIdx.x << 10));
    __shared__ float red[256];
    const int t = threadIdx.x;
    float4 x = pr[t];
    float mx = fmaxf(fmaxf(x.x, x.y), fmaxf(x.z, x.w));
    red[t] = mx; __syncthreads();
    for (int s = 128; s > 0; s >>= 1) { if (t < s) red[t] = fmaxf(red[t], red[t + s]); __syncthreads(); }
    mx = red[0]; __syncthreads();
    x.x = expf(x.x - mx); x.y = expf(x.y - mx); x.z = expf(x.z - mx); x.w = expf(x.w - mx);
    red[t] = x.x + x.y + x.z + x.w; __syncthreads();
    for (int s = 128; s > 0; s >>= 1) { if (t < s) red[t] += red[t + s]; __syncthreads(); }
    const float inv = 1.0f / red[0];
    x.x *= inv; x.y *= inv; x.z *= inv; x.w *= inv;
    pr[t] = x;
}

// Row LayerNorm over 1024 columns (one block per row); optional bf16 copy.
__global__ __launch_bounds__(256) void k_layernorm(
    const float* __restrict__ in, const float* __restrict__ g,
    const float* __restrict__ be, float* __restrict__ outF,
    unsigned short* __restrict__ outB)
{
    const long rowoff = (long)blockIdx.x << 10;
    const float4* pr = (const float4*)(in + rowoff);
    __shared__ float r1[256], r2[256];
    const int t = threadIdx.x;
    float4 x = pr[t];
    r1[t] = x.x + x.y + x.z + x.w;
    r2[t] = x.x * x.x + x.y * x.y + x.z * x.z + x.w * x.w;
    __syncthreads();
    for (int s = 128; s > 0; s >>= 1) {
        if (t < s) { r1[t] += r1[t + s]; r2[t] += r2[t + s]; }
        __syncthreads();
    }
    const float mean = r1[0] * (1.0f / DD);
    const float var  = r2[0] * (1.0f / DD) - mean * mean;
    const float rstd = rsqrtf(var + 1e-12f);
    const int c = t * 4;
    float y0 = g[c + 0] * (x.x - mean) * rstd + be[c + 0];
    float y1 = g[c + 1] * (x.y - mean) * rstd + be[c + 1];
    float y2 = g[c + 2] * (x.z - mean) * rstd + be[c + 2];
    float y3 = g[c + 3] * (x.w - mean) * rstd + be[c + 3];
    float4* po = (float4*)(outF + rowoff);
    po[t] = make_float4(y0, y1, y2, y3);
    if (outB) {
        unsigned short* pb = outB + rowoff;
        pb[c + 0] = f2bf(y0); pb[c + 1] = f2bf(y1);
        pb[c + 2] = f2bf(y2); pb[c + 3] = f2bf(y3);
    }
}

// ---------------------------------------------------------------------------
// Host-side launch
// ---------------------------------------------------------------------------
extern "C" void kernel_launch(void* const* d_in, const int* in_sizes, int n_in,
                              void* d_out, int out_size, void* d_ws, size_t ws_size,
                              hipStream_t stream)
{
    const float* x    = (const float*)d_in[0];
    const int*   mask = (const int*)d_in[1];
    const float* Wq   = (const float*)d_in[2];
    const float* bq   = (const float*)d_in[3];
    const float* Wo   = (const float*)d_in[4];
    const float* bo   = (const float*)d_in[5];
    const float* W1   = (const float*)d_in[6];
    const float* b1   = (const float*)d_in[7];
    const float* W2   = (const float*)d_in[8];
    const float* b2   = (const float*)d_in[9];
    const float* g1   = (const float*)d_in[10];
    const float* be1  = (const float*)d_in[11];
    const float* g2   = (const float*)d_in[12];
    const float* be2  = (const float*)d_in[13];

    float* out2 = (float*)d_out;                       // [B,S,D]
    float* attn = (float*)d_out + OUT2_ELEMS;          // [B,H,S,S]

    // Workspace layout (bytes)
    char* ws = (char*)d_ws;
    const size_t MB = 1ull << 20;
    unsigned short* WqT    = (unsigned short*)(ws + 0 * MB);     // [1024][1024]   2 MB
    unsigned short* WoT    = (unsigned short*)(ws + 2 * MB);     // [1024][1024]   2 MB
    unsigned short* W1T    = (unsigned short*)(ws + 4 * MB);     // [4096][1024]   8 MB
    unsigned short* W2T    = (unsigned short*)(ws + 12 * MB);    // [1024][4096]   8 MB
    unsigned short* Pbf    = (unsigned short*)(ws + 20 * MB);    // [4096][1024]   8 MB
    unsigned short* Vt     = (unsigned short*)(ws + 28 * MB);    // [B*H][64][1024] 8 MB
    unsigned short* AObf   = (unsigned short*)(ws + 36 * MB);    // [4096][1024]   8 MB
    unsigned short* out1bf = (unsigned short*)(ws + 44 * MB);    // [4096][1024]   8 MB
    unsigned short* ff1bf  = (unsigned short*)(ws + 52 * MB);    // [4096][4096]  32 MB
    float*          res1   = (float*)(ws + 84 * MB);             // [4096][1024]  16 MB
    float*          out1   = (float*)(ws + 100 * MB);            // [4096][1024]  16 MB
    float*          res2   = (float*)(ws + 116 * MB);            // [4096][1024]  16 MB

    // 1) Weight convert + transpose to bf16 [N][K]
    k_conv_transpose<<<(1024 * 1024 + 255) / 256, 256, 0, stream>>>(Wq, WqT, DD, DD);
    k_conv_transpose<<<(1024 * 1024 + 255) / 256, 256, 0, stream>>>(Wo, WoT, DD, DD);
    k_conv_transpose<<<(4096 * 1024 + 255) / 256, 256, 0, stream>>>(W1, W1T, DD, DFF_);
    k_conv_transpose<<<(4096 * 1024 + 255) / 256, 256, 0, stream>>>(W2, W2T, DFF_, DD);

    // 2) P = x @ Wq + bq  (qkv shared projection), store bf16
    gemm_wmma<true, true, false, false, false, false, true, false, false, false>
        <<<dim3(DD / 64, MTOT / 256, 1), 256, 0, stream>>>(
            x, WqT, nullptr, Pbf, bq, nullptr, nullptr, 0.0f,
            MTOT, DD, DD, DD, DD, DD, 0, 0, 0, 1);

    // 3) Per-head V^T for the attn@V GEMM
    k_build_vt<<<(BB * HH * HD_ * SS_ + 255) / 256, 256, 0, stream>>>(Pbf, Vt);

    // 4) scores = q k^T / sqrt(64), mask -> f32 into d_out attn region
    gemm_wmma<false, false, false, false, true, true, false, true, true, false>
        <<<dim3(SS_ / 64, SS_ / 256, BB * HH), 256, 0, stream>>>(
            Pbf, Pbf, attn, nullptr, nullptr, nullptr, mask, 0.125f,
            SS_, SS_, HD_, DD, DD, SS_,
            0, 0, (long)SS_ * SS_, HH);

    // 5) row softmax, in place
    k_softmax<<<BB * HH * SS_, 256, 0, stream>>>(attn);

    // 6) attn_out(head) = probs @ v  -> bf16 written at [b][s][h*64+d]
    gemm_wmma<true, false, false, false, false, false, true, false, false, true>
        <<<dim3(HD_ / 64, SS_ / 256, BB * HH), 256, 0, stream>>>(
            attn, Vt, nullptr, AObf, nullptr, nullptr, nullptr, 0.0f,
            SS_, HD_, SS_, SS_, SS_, DD,
            (long)SS_ * SS_, (long)HD_ * SS_, 0, HH);

    // 7) res1 = attn_out @ Wo + bo + x
    gemm_wmma<false, true, true, false, false, true, false, false, false, false>
        <<<dim3(DD / 64, MTOT / 256, 1), 256, 0, stream>>>(
            AObf, WoT, res1, nullptr, bo, x, nullptr, 0.0f,
            MTOT, DD, DD, DD, DD, DD, 0, 0, 0, 1);

    // 8) out1 = LayerNorm(res1)
    k_layernorm<<<MTOT, 256, 0, stream>>>(res1, g1, be1, out1, out1bf);

    // 9) ff1 = relu(out1 @ W1 + b1), bf16
    gemm_wmma<false, true, false, true, false, false, true, false, false, false>
        <<<dim3(DFF_ / 64, MTOT / 256, 1), 256, 0, stream>>>(
            out1bf, W1T, nullptr, ff1bf, b1, nullptr, nullptr, 0.0f,
            MTOT, DFF_, DD, DD, DD, DFF_, 0, 0, 0, 1);

    // 10) res2 = ff1 @ W2 + b2 + out1
    gemm_wmma<false, true, true, false, false, true, false, false, false, false>
        <<<dim3(DD / 64, MTOT / 256, 1), 256, 0, stream>>>(
            ff1bf, W2T, res2, nullptr, b2, out1, nullptr, 0.0f,
            MTOT, DD, DFF_, DFF_, DFF_, DD, 0, 0, 0, 1);

    // 11) out2 = LayerNorm(res2) -> d_out
    k_layernorm<<<MTOT, 256, 0, stream>>>(res2, g2, be2, out2, nullptr);
}
